// ValueNetwork_57114475102901
// MI455X (gfx1250) — matrix operand order repaired and checked by reading, not compiled, not run on hardware
//
#include <hip/hip_runtime.h>
#include <hip/hip_bf16.h>
#include <math.h>

typedef __attribute__((ext_vector_type(16))) _Float16 v16h;
typedef __attribute__((ext_vector_type(8)))  float    v8f;

// ---------------------------------------------------------------------------
// Fragment helpers (wave32, 16x16x32 f16 WMMA, layouts per CDNA5 ISA)
// ---------------------------------------------------------------------------

// A fragment: 16x32 f16 tile from row-major LDS buffer (stride ld halves).
// lane 0-15 : row m0+lane,  K = k0+0..7  (elems 0-7) and k0+16..23 (elems 8-15)
// lane 16-31: row m0+l-16,  K = k0+8..15 (elems 0-7) and k0+24..31 (elems 8-15)
__device__ __forceinline__ v16h load_a(const _Float16* S, int ld, int m0, int k0) {
    int lane = threadIdx.x & 31;
    int m  = m0 + (lane & 15);
    int kh = k0 + ((lane >> 4) << 3);
    const _Float16* p = S + m * ld + kh;
    v16h a;
#pragma unroll
    for (int i = 0; i < 8; ++i) { a[i] = p[i]; a[8 + i] = p[16 + i]; }
    return a;
}

// B fragment: pre-packed in global memory, 512 halves per 32x16 block,
// laid out [lane][elem] so each lane reads 32 contiguous bytes.
__device__ __forceinline__ v16h load_b(const _Float16* P) {
    int lane = threadIdx.x & 31;
    const _Float16* p = P + lane * 16;
    v16h b;
#pragma unroll
    for (int i = 0; i < 16; ++i) b[i] = p[i];
    return b;
}

// D/C fragment store: lane 0-15 -> col n0+lane rows m0+0..7 ; lane16-31 -> rows m0+8..15
__device__ __forceinline__ void store_d(_Float16* S, int ld, int m0, int n0,
                                        v8f c, float bias, bool do_relu) {
    int lane = threadIdx.x & 31;
    int n  = n0 + (lane & 15);
    int mb = m0 + ((lane >> 4) << 3);
#pragma unroll
    for (int v = 0; v < 8; ++v) {
        float x = c[v] + bias;
        if (do_relu) x = fmaxf(x, 0.0f);
        S[(mb + v) * ld + n] = (_Float16)x;
    }
}

// 64 x Nout = (64 x K) @ (K x Nout), 8 waves cooperate. K,Nout multiples of 32/16.
// Tile counts (16/32/64) are multiples of 8 -> every wave fully active at WMMA.
__device__ __forceinline__ void gemm64(const _Float16* SA, int lda,
                                       _Float16* SD, int ldd,
                                       const _Float16* Wp, const float* bias,
                                       int K, int Nout, bool do_relu) {
    int wave  = threadIdx.x >> 5;
    int ktn   = K >> 5;
    int tiles = (Nout >> 4) << 2;   // n-tiles * 4 m-tiles
    for (int t = wave; t < tiles; t += 8) {
        int nt = t >> 2;
        int mt = t & 3;
        v8f c = {};
        const _Float16* Pb = Wp + (size_t)nt * ktn * 512;
#pragma unroll 4
        for (int kt = 0; kt < ktn; ++kt) {
            v16h a = load_a(SA, lda, mt * 16, kt * 32);
            v16h b = load_b(Pb + kt * 512);
            c = __builtin_amdgcn_wmma_f32_16x16x32_f16(false, a, false, b,
                                                       (short)0, c, false, false);
        }
        float bv = bias[nt * 16 + (threadIdx.x & 15)];
        store_d(SD, ldd, mt * 16, nt * 16, c, bv, do_relu);
    }
}

// ---------------------------------------------------------------------------
// Weight pack: f32 row-major (Ksrc x N) -> f16 WMMA B-fragment blocks,
// K zero-padded to Kpad (multiple of 32).
// ---------------------------------------------------------------------------
__global__ void k_pack(const float* __restrict__ W, _Float16* __restrict__ P,
                       int Ksrc, int Kpad, int N) {
    int idx = blockIdx.x * blockDim.x + threadIdx.x;
    int total = Kpad * N;
    if (idx >= total) return;
    int e    = idx & 15;
    int lane = (idx >> 4) & 31;
    int blk  = idx >> 9;
    int ktn  = Kpad >> 5;
    int kt   = blk % ktn;
    int nt   = blk / ktn;
    int n = nt * 16 + (lane & 15);
    int k = kt * 32 + ((lane >> 4) << 3) + (e & 7) + ((e >> 3) << 4);
    float v = (k < Ksrc) ? W[(size_t)k * N + n] : 0.0f;
    P[idx] = (_Float16)v;
}

// ---------------------------------------------------------------------------
// Main kernel: one workgroup per batch element (B=4096), 256 threads.
// Full per-row pipeline + mean pool + attention softmax + weighted sum.
// Writes padded joint row [self(9) | weighted(128) | 0-pad(23)] (160 halves).
// ---------------------------------------------------------------------------
#define LDA 264   // 256+8 halves: conflict-free fragment loads
#define LDM 136   // 128+8
__global__ void __launch_bounds__(256)
k_main(const float* __restrict__ state,
       const _Float16* pw1a, const float* b1a,
       const _Float16* pw1b, const float* b1b,
       const _Float16* pw2a, const float* b2a,
       const _Float16* pw2b, const float* b2b,
       const _Float16* pwa1, const float* ba1,
       const _Float16* pwa2, const float* ba2,
       const float* wa3, const float* ba3,
       _Float16* __restrict__ joint) {
    __shared__ _Float16 sA[64 * LDA];
    __shared__ _Float16 sB[64 * LDA];
    __shared__ _Float16 sM2[64 * LDM];
    __shared__ float sGs[128];
    __shared__ float sSc[64];
    __shared__ float sInv;

    int b = blockIdx.x;
    const float* xb = state + (size_t)b * 64 * 64;

    // stage x (64x64 f32 -> f16)
    for (int i = threadIdx.x; i < 64 * 64; i += 256) {
        int r = i >> 6, c = i & 63;
        sA[r * LDA + c] = (_Float16)xb[i];
    }
    __syncthreads();
    gemm64(sA, LDA, sB, LDA, pw1a, b1a, 64, 256, true);    // h1 = relu(x@w1a+b)
    __syncthreads();
    gemm64(sB, LDA, sA, LDA, pw1b, b1b, 256, 128, true);   // m1 = relu(h1@w1b+b)
    __syncthreads();
    gemm64(sA, LDA, sB, LDA, pw2a, b2a, 128, 256, true);   // h2 = relu(m1@w2a+b)
    __syncthreads();
    gemm64(sB, LDA, sM2, LDM, pw2b, b2b, 256, 128, false); // m2 = h2@w2b+b (unmasked m1 used)
    __syncthreads();

    // visibility mask: vis[r] = state[b,r,61] > 0 ; mask m1 and m2
    for (int i = threadIdx.x; i < 64 * 128; i += 256) {
        int r = i >> 7, c = i & 127;
        if (!(xb[r * 64 + 61] > 0.0f)) {
            sA[r * LDA + c]  = (_Float16)0.0f;
            sM2[r * LDM + c] = (_Float16)0.0f;
        }
    }
    __syncthreads();

    // gs = mean over rows of masked m1
    if (threadIdx.x < 128) {
        float s = 0.0f;
        for (int r = 0; r < 64; ++r) s += (float)sA[r * LDA + threadIdx.x];
        sGs[threadIdx.x] = s * (1.0f / 64.0f);
    }
    __syncthreads();

    // att_in = [m1 | gs] into sB (64 x 256)
    for (int i = threadIdx.x; i < 64 * 256; i += 256) {
        int r = i >> 8, c = i & 255;
        sB[r * LDA + c] = (c < 128) ? sA[r * LDA + c] : (_Float16)sGs[c - 128];
    }
    __syncthreads();
    gemm64(sB, LDA, sA, LDA, pwa1, ba1, 256, 128, true);   // a1
    __syncthreads();
    gemm64(sA, LDA, sB, LDA, pwa2, ba2, 128, 64, true);    // a2
    __syncthreads();

    // scores -> gated exp
    if (threadIdx.x < 64) {
        float s = ba3[0];
        for (int c = 0; c < 64; ++c) s += (float)sB[threadIdx.x * LDA + c] * wa3[c];
        sSc[threadIdx.x] = (s != 0.0f) ? expf(s) : 0.0f;
    }
    __syncthreads();
    if (threadIdx.x == 0) {
        float t = 0.0f;
        for (int n = 0; n < 64; ++n) t += sSc[n];
        sInv = 1.0f / t;
    }
    __syncthreads();
    float inv = sInv;

    _Float16* jrow = joint + (size_t)b * 160;
    if (threadIdx.x < 128) {
        int c = threadIdx.x;
        float acc = 0.0f;
        for (int n = 0; n < 64; ++n) acc += (sSc[n] * inv) * (float)sM2[n * LDM + c];
        jrow[9 + c] = (_Float16)acc;
    } else {
        int j = threadIdx.x - 128;
        if (j < 9)       jrow[j] = (_Float16)xb[j];             // self_state = state[b,0,0:9]
        else if (j < 32) jrow[137 + (j - 9)] = (_Float16)0.0f;  // zero pad 137..159
    }
}

// ---------------------------------------------------------------------------
// Head kernel: value = relu(relu(joint@w3a+b)@w3b+b)@w3c+b over 4096 rows.
// 64 rows per workgroup. K(joint) padded to 160.
// ---------------------------------------------------------------------------
#define LDJ 168   // 160+8
__global__ void __launch_bounds__(256)
k_head(const _Float16* __restrict__ joint,
       const _Float16* pw3a, const float* b3a,
       const _Float16* pw3b, const float* b3b,
       const float* w3c, const float* b3c,
       float* __restrict__ out) {
    __shared__ _Float16 sA[64 * LDJ];
    __shared__ _Float16 sB[64 * LDA];
    int rb = blockIdx.x * 64;

    for (int i = threadIdx.x; i < 64 * 160; i += 256) {
        int r = i / 160, c = i - r * 160;
        sA[r * LDJ + c] = joint[(size_t)(rb + r) * 160 + c];
    }
    __syncthreads();
    gemm64(sA, LDJ, sB, LDA, pw3a, b3a, 160, 256, true);
    __syncthreads();
    gemm64(sB, LDA, sA, LDJ, pw3b, b3b, 256, 128, true);
    __syncthreads();
    if (threadIdx.x < 64) {
        int r = threadIdx.x;
        float s = b3c[0];
        for (int c = 0; c < 128; ++c) s += (float)sA[r * LDJ + c] * w3c[c];
        out[rb + r] = s;
    }
}

// ---------------------------------------------------------------------------
// Launch
// ---------------------------------------------------------------------------
extern "C" void kernel_launch(void* const* d_in, const int* in_sizes, int n_in,
                              void* d_out, int out_size, void* d_ws, size_t ws_size,
                              hipStream_t stream) {
    const float* state = (const float*)d_in[0];
    const float* w1a = (const float*)d_in[1];  const float* b1a = (const float*)d_in[2];
    const float* w1b = (const float*)d_in[3];  const float* b1b = (const float*)d_in[4];
    const float* w2a = (const float*)d_in[5];  const float* b2a = (const float*)d_in[6];
    const float* w2b = (const float*)d_in[7];  const float* b2b = (const float*)d_in[8];
    const float* wa1 = (const float*)d_in[9];  const float* ba1 = (const float*)d_in[10];
    const float* wa2 = (const float*)d_in[11]; const float* ba2 = (const float*)d_in[12];
    const float* wa3 = (const float*)d_in[13]; const float* ba3 = (const float*)d_in[14];
    const float* w3a = (const float*)d_in[15]; const float* b3a = (const float*)d_in[16];
    const float* w3b = (const float*)d_in[17]; const float* b3b = (const float*)d_in[18];
    const float* w3c = (const float*)d_in[19]; const float* b3c = (const float*)d_in[20];
    float* out = (float*)d_out;

    _Float16* ws = (_Float16*)d_ws;
    size_t off = 0;
    _Float16* p_w1a = ws + off; off += 64  * 256;   // Kpad x N halves
    _Float16* p_w1b = ws + off; off += 256 * 128;
    _Float16* p_w2a = ws + off; off += 128 * 256;
    _Float16* p_w2b = ws + off; off += 256 * 128;
    _Float16* p_wa1 = ws + off; off += 256 * 128;
    _Float16* p_wa2 = ws + off; off += 128 * 64;
    _Float16* p_w3a = ws + off; off += 160 * 256;
    _Float16* p_w3b = ws + off; off += 256 * 128;
    _Float16* joint = ws + off; off += (size_t)4096 * 160;

    auto pack = [&](const float* W, _Float16* P, int Ksrc, int Kpad, int N) {
        int total = Kpad * N;
        k_pack<<<(total + 255) / 256, 256, 0, stream>>>(W, P, Ksrc, Kpad, N);
    };
    pack(w1a, p_w1a,  64,  64, 256);
    pack(w1b, p_w1b, 256, 256, 128);
    pack(w2a, p_w2a, 128, 128, 256);
    pack(w2b, p_w2b, 256, 256, 128);
    pack(wa1, p_wa1, 256, 256, 128);
    pack(wa2, p_wa2, 128, 128,  64);
    pack(w3a, p_w3a, 137, 160, 256);
    pack(w3b, p_w3b, 256, 256, 128);

    k_main<<<4096, 256, 0, stream>>>(state,
                                     p_w1a, b1a, p_w1b, b1b,
                                     p_w2a, b2a, p_w2b, b2b,
                                     p_wa1, ba1, p_wa2, ba2,
                                     wa3, ba3, joint);
    k_head<<<64, 256, 0, stream>>>(joint, p_w3a, b3a, p_w3b, b3b, w3c, b3c, out);
}
